// wconv_3x3_63041529970868
// MI455X (gfx1250) — matrix operand order repaired and checked by reading, not compiled, not run on hardware
//
#include <hip/hip_runtime.h>

// Winograd F(2x2,3x3): N=16, C=256, K=256, H=W=56
#define NB 16
#define CH 256
#define KF 256
#define HW 56
#define OH 28
#define OW 28
#define PT (NB*OH*OW)          // 12544 tiles
#define LDS_STRIDE 40          // halves per padded LDS row (80B, 16B-aligned)

typedef __attribute__((ext_vector_type(8)))  float  v8f;
typedef __attribute__((ext_vector_type(16))) __bf16 v16bf;
typedef __attribute__((ext_vector_type(4)))  int    v4i;

union FragB16 { v16bf v; float4 q[2]; };
union F2U4    { float2 f; unsigned short s[4]; };

#define AS1 __attribute__((address_space(1)))
#define AS3 __attribute__((address_space(3)))

#if __has_builtin(__builtin_amdgcn_global_load_async_to_lds_b128) && \
    __has_builtin(__builtin_amdgcn_s_wait_asynccnt)
#define USE_ASYNC_LDS 1
#endif

__device__ __forceinline__ void g2l_b128(const unsigned short* g, unsigned short* l) {
#if defined(USE_ASYNC_LDS)
    __builtin_amdgcn_global_load_async_to_lds_b128(
        (AS1 v4i*)(__UINTPTR_TYPE__)g,
        (AS3 v4i*)(unsigned)(__UINTPTR_TYPE__)l,
        0, 0);
#else
    *(float4*)l = *(const float4*)g;
#endif
}

#if defined(USE_ASYNC_LDS)
#define WAIT_ASYNC(n) __builtin_amdgcn_s_wait_asynccnt(n)
#else
#define WAIT_ASYNC(n)
#endif

__device__ __forceinline__ unsigned short f2bf_rne(float x) {
    unsigned u = __float_as_uint(x);
    unsigned r = u + 0x7fffu + ((u >> 16) & 1u);   // round-to-nearest-even
    return (unsigned short)(r >> 16);
}

// ---------------------------------------------------------------------------
// Kernel 1: filter transform  U[ab][k][c] = (G g G^T)[ab], bf16  (c contiguous)
// ---------------------------------------------------------------------------
__global__ __launch_bounds__(256)
void wg_filter_transform(const float* __restrict__ w, unsigned short* __restrict__ U) {
    int idx = blockIdx.x * 256 + threadIdx.x;     // k*C + c
    int k = idx >> 8;
    int c = idx & 255;
    const float* g = w + (size_t)(k * CH + c) * 9;
    float g00=g[0],g01=g[1],g02=g[2];
    float g10=g[3],g11=g[4],g12=g[5];
    float g20=g[6],g21=g[7],g22=g[8];

    float t[4][3];
    t[0][0]=g00;                 t[0][1]=g01;                 t[0][2]=g02;
    t[1][0]=0.5f*(g00+g10+g20);  t[1][1]=0.5f*(g01+g11+g21);  t[1][2]=0.5f*(g02+g12+g22);
    t[2][0]=0.5f*(g00-g10+g20);  t[2][1]=0.5f*(g01-g11+g21);  t[2][2]=0.5f*(g02-g12+g22);
    t[3][0]=g20;                 t[3][1]=g21;                 t[3][2]=g22;

    #pragma unroll
    for (int i = 0; i < 4; ++i) {
        float u0 = t[i][0];
        float u1 = 0.5f*(t[i][0]+t[i][1]+t[i][2]);
        float u2 = 0.5f*(t[i][0]-t[i][1]+t[i][2]);
        float u3 = t[i][2];
        size_t base = (size_t)(i*4) * KF * CH + (size_t)k * CH + c;
        U[base + 0*(size_t)KF*CH] = f2bf_rne(u0);
        U[base + 1*(size_t)KF*CH] = f2bf_rne(u1);
        U[base + 2*(size_t)KF*CH] = f2bf_rne(u2);
        U[base + 3*(size_t)KF*CH] = f2bf_rne(u3);
    }
}

// ---------------------------------------------------------------------------
// Kernel 2: input transform -> V2[ab][p][c] bf16 (c contiguous), LDS transpose
// Block: 32c x 32p tile, 256 threads (each does 4 c values at one p).
// ---------------------------------------------------------------------------
__global__ __launch_bounds__(256)
void wg_input_transform(const float* __restrict__ x, unsigned short* __restrict__ V2) {
    __shared__ __align__(16) unsigned short buf[16 * 32 * LDS_STRIDE]; // [ab][p][c pad 40]

    const int tid = threadIdx.x;
    const int c0 = blockIdx.x * 32;
    const int p0 = blockIdx.y * 32;
    const int pl = tid & 31;          // p within tile
    const int cb = (tid >> 5) * 4;    // first of 4 c values

    const int p  = p0 + pl;
    const int n  = p / (OH*OW);
    const int r  = p - n * (OH*OW);
    const int th = r / OW;
    const int tw = r - th * OW;
    const int ih0 = th*2 - 1, iw0 = tw*2 - 1;

    unsigned short acc[16][4];

    #pragma unroll
    for (int i = 0; i < 4; ++i) {
        const int c = c0 + cb + i;
        const float* xb = x + ((size_t)(n * CH + c) * HW) * HW;

        float d[4][4];
        #pragma unroll
        for (int j = 0; j < 4; ++j) {
            int ih = ih0 + j;
            bool okh = (ih >= 0) && (ih < HW);
            #pragma unroll
            for (int l = 0; l < 4; ++l) {
                int iw = iw0 + l;
                bool ok = okh && (iw >= 0) && (iw < HW);
                d[j][l] = ok ? xb[ih*HW + iw] : 0.0f;
            }
        }

        float t0[4], t1[4], t2[4], t3[4];
        #pragma unroll
        for (int l = 0; l < 4; ++l) {
            t0[l] = d[0][l] - d[2][l];
            t1[l] = d[1][l] + d[2][l];
            t2[l] = d[2][l] - d[1][l];
            t3[l] = d[1][l] - d[3][l];
        }
        float vv[4][4];
        #pragma unroll
        for (int ii = 0; ii < 4; ++ii) {
            const float* ti = (ii==0)?t0:(ii==1)?t1:(ii==2)?t2:t3;
            vv[ii][0] = ti[0] - ti[2];
            vv[ii][1] = ti[1] + ti[2];
            vv[ii][2] = ti[2] - ti[1];
            vv[ii][3] = ti[1] - ti[3];
        }
        #pragma unroll
        for (int ii = 0; ii < 4; ++ii)
            #pragma unroll
            for (int mm = 0; mm < 4; ++mm)
                acc[ii*4 + mm][i] = f2bf_rne(vv[ii][mm]);
    }

    // packed b64 stores into LDS: buf[ab][pl][cb..cb+3]
    #pragma unroll
    for (int ab = 0; ab < 16; ++ab) {
        F2U4 t;
        t.s[0] = acc[ab][0]; t.s[1] = acc[ab][1];
        t.s[2] = acc[ab][2]; t.s[3] = acc[ab][3];
        *(float2*)&buf[(ab*32 + pl) * LDS_STRIDE + cb] = t.f;
    }
    __syncthreads();

    // coalesced output: 512 rows of 64B ((ab,p) row over c)
    #pragma unroll
    for (int j = 0; j < 2; ++j) {
        const int row = tid + j*256;       // 0..511
        const int ab  = row >> 5;
        const int pp  = row & 31;
        const unsigned short* src = &buf[(ab*32 + pp) * LDS_STRIDE];
        unsigned short* dst = V2 + ((size_t)ab * PT + (p0 + pp)) * CH + c0;
        *(float4*)&dst[0]  = *(const float4*)&src[0];
        *(float4*)&dst[8]  = *(const float4*)&src[8];
        *(float4*)&dst[16] = *(const float4*)&src[16];
        *(float4*)&dst[24] = *(const float4*)&src[24];
    }
}

// ---------------------------------------------------------------------------
// Kernel 3: fused 16x batched GEMM (bf16 WMMA, fp32 acc) + output transform
// Block: 256 threads = 8 waves; tile = 128k x 64p; 2x2 subtiles per wave.
// Double-buffered async global->LDS staging; loop unroll capped at 2 to stay
// under 256 VGPRs (no vgpr-msb mov traffic).
// ---------------------------------------------------------------------------
__global__ __launch_bounds__(256)
void wg_gemm_fused(const unsigned short* __restrict__ U,
                   const unsigned short* __restrict__ V2,
                   float* __restrict__ Y) {
    __shared__ __align__(16) unsigned short Us[2][128 * LDS_STRIDE]; // 128k x 32c
    __shared__ __align__(16) unsigned short Vs[2][ 64 * LDS_STRIDE]; //  64p x 32c

    const int tid  = threadIdx.x;
    const int lane = tid & 31;
    const int wave = tid >> 5;
    const int k0 = blockIdx.x * 128;
    const int p0 = blockIdx.y * 64;
    const int skw = wave >> 1;          // 0..3: 32-k slice
    const int sp0 = (wave & 1) * 2;     // 0 or 2: first of two p-subtiles

    const int arow0 = skw*32 + (lane & 15);
    const int arow1 = arow0 + 16;
    const int ag0   = (lane >> 4) * 8;
    const int brow  = sp0*16 + (lane & 15);
    const int bg0   = (lane >> 4) * 8;

    const int vrow = tid >> 2;          // 0..63 V staging row
    const int vseg = (tid & 3) * 8;     // halves

    v8f Yacc[2][2][2][2];               // [ak][bp][u][v]
    #pragma unroll
    for (int ak = 0; ak < 2; ++ak)
        #pragma unroll
        for (int bp = 0; bp < 2; ++bp)
            #pragma unroll
            for (int u = 0; u < 2; ++u)
                #pragma unroll
                for (int v = 0; v < 2; ++v)
                    Yacc[ak][bp][u][v] = (v8f){0,0,0,0,0,0,0,0};
    v8f macc[2][2];
    #pragma unroll
    for (int ak = 0; ak < 2; ++ak)
        #pragma unroll
        for (int bp = 0; bp < 2; ++bp)
            macc[ak][bp] = (v8f){0,0,0,0,0,0,0,0};

    // stage 0
    #pragma unroll
    for (int j = 0; j < 2; ++j) {
        const int id = tid + 256*j;
        const int urow = id >> 2, useg = (id & 3) * 8;
        g2l_b128(U + ((size_t)(k0 + urow))*CH + useg,
                 &Us[0][urow*LDS_STRIDE + useg]);
    }
    g2l_b128(V2 + ((size_t)(p0 + vrow))*CH + vseg,
             &Vs[0][vrow*LDS_STRIDE + vseg]);

    #pragma unroll 2
    for (int s = 0; s < 128; ++s) {     // s = ab*8 + chunk
        const int bufc = s & 1;
        if (s + 1 < 128) {
            const int abn = (s+1) >> 3;
            const int ccn = ((s+1) & 7) << 5;
            #pragma unroll
            for (int j = 0; j < 2; ++j) {
                const int id = tid + 256*j;
                const int urow = id >> 2, useg = (id & 3) * 8;
                g2l_b128(U + ((size_t)abn*KF + k0 + urow)*CH + ccn + useg,
                         &Us[bufc^1][urow*LDS_STRIDE + useg]);
            }
            g2l_b128(V2 + ((size_t)abn*PT + p0 + vrow)*CH + ccn + vseg,
                     &Vs[bufc^1][vrow*LDS_STRIDE + vseg]);
            WAIT_ASYNC(3);              // in-order: stage s finished
        } else {
            WAIT_ASYNC(0);
        }
        __syncthreads();

        FragB16 a0, a1, b0, b1;
        const unsigned short* us = &Us[bufc][0];
        const unsigned short* vs = &Vs[bufc][0];
        a0.q[0] = *(const float4*)&us[arow0*LDS_STRIDE + ag0];
        a0.q[1] = *(const float4*)&us[arow0*LDS_STRIDE + 16 + ag0];
        a1.q[0] = *(const float4*)&us[arow1*LDS_STRIDE + ag0];
        a1.q[1] = *(const float4*)&us[arow1*LDS_STRIDE + 16 + ag0];
        b0.q[0] = *(const float4*)&vs[brow*LDS_STRIDE + bg0];
        b0.q[1] = *(const float4*)&vs[brow*LDS_STRIDE + bg0 + 16];
        b1.q[0] = *(const float4*)&vs[(brow + 16)*LDS_STRIDE + bg0];
        b1.q[1] = *(const float4*)&vs[(brow + 16)*LDS_STRIDE + bg0 + 16];

        macc[0][0] = __builtin_amdgcn_wmma_f32_16x16x32_bf16(
                         false, a0.v, false, b0.v, (short)0, macc[0][0], false, false);
        macc[0][1] = __builtin_amdgcn_wmma_f32_16x16x32_bf16(
                         false, a0.v, false, b1.v, (short)0, macc[0][1], false, false);
        macc[1][0] = __builtin_amdgcn_wmma_f32_16x16x32_bf16(
                         false, a1.v, false, b0.v, (short)0, macc[1][0], false, false);
        macc[1][1] = __builtin_amdgcn_wmma_f32_16x16x32_bf16(
                         false, a1.v, false, b1.v, (short)0, macc[1][1], false, false);

        __syncthreads();

        if ((s & 7) == 7) {             // end of ab-plane: fold M into Y
            const int a = (s >> 5) & 3;          // ab = s>>3; a = ab>>2
            const int b = (s >> 3) & 3;
            // branchless AT coefficients: AT0 = {1,1,1,0}; AT1 = {0,1,-1,-1}
            const float a0c = (a == 3) ? 0.f : 1.f;
            const float a1c = (a == 0) ? 0.f : ((a == 1) ? 1.f : -1.f);
            const float b0c = (b == 3) ? 0.f : 1.f;
            const float b1c = (b == 0) ? 0.f : ((b == 1) ? 1.f : -1.f);
            const float c00 = a0c * b0c;
            const float c01 = a0c * b1c;
            const float c10 = a1c * b0c;
            const float c11 = a1c * b1c;
            #pragma unroll
            for (int ak = 0; ak < 2; ++ak) {
                #pragma unroll
                for (int bp = 0; bp < 2; ++bp) {
                    #pragma unroll
                    for (int j = 0; j < 8; ++j) {
                        float m = macc[ak][bp][j];
                        Yacc[ak][bp][0][0][j] += c00 * m;
                        Yacc[ak][bp][0][1][j] += c01 * m;
                        Yacc[ak][bp][1][0][j] += c10 * m;
                        Yacc[ak][bp][1][1][j] += c11 * m;
                    }
                    macc[ak][bp] = (v8f){0,0,0,0,0,0,0,0};
                }
            }
        }
    }

    // epilogue: 2x2 outputs per (k,p) element, float2 (v-pair) stores
    #pragma unroll
    for (int ak = 0; ak < 2; ++ak) {
        #pragma unroll
        for (int bp = 0; bp < 2; ++bp) {
            const int p  = p0 + (sp0 + bp)*16 + (lane & 15);
            const int n  = p / (OH*OW);
            const int r  = p - n * (OH*OW);
            const int th = r / OW;
            const int tw = r - th * OW;
            #pragma unroll
            for (int j = 0; j < 8; ++j) {
                const int k = k0 + skw*32 + ak*16 + j + ((lane >> 4) * 8);
                #pragma unroll
                for (int u = 0; u < 2; ++u) {
                    float2 val;
                    val.x = Yacc[ak][bp][u][0][j];
                    val.y = Yacc[ak][bp][u][1][j];
                    *(float2*)&Y[(((size_t)n * KF + k) * HW + (2*th + u)) * HW + 2*tw] = val;
                }
            }
        }
    }
}

// ---------------------------------------------------------------------------
extern "C" void kernel_launch(void* const* d_in, const int* in_sizes, int n_in,
                              void* d_out, int out_size, void* d_ws, size_t ws_size,
                              hipStream_t stream) {
    const float* x = (const float*)d_in[0];     // (N,C,H,W) fp32
    const float* w = (const float*)d_in[1];     // (K,C,3,3) fp32
    float* y = (float*)d_out;                   // (N,K,56,56) fp32

    unsigned short* U  = (unsigned short*)d_ws;                                // 2 MB
    unsigned short* V2 = (unsigned short*)((char*)d_ws + (size_t)2*1024*1024); // ~103 MB

    wg_filter_transform<<<(KF * CH) / 256, 256, 0, stream>>>(w, U);

    dim3 g2(CH / 32, PT / 32);
    wg_input_transform<<<g2, 256, 0, stream>>>(x, V2);

    dim3 g3(KF / 128, PT / 64);
    wg_gemm_fused<<<g3, 256, 0, stream>>>(U, V2, y);
}
